// UNet_46995532153388
// MI455X (gfx1250) — matrix operand order repaired and checked by reading, not compiled, not run on hardware
//
#include <hip/hip_runtime.h>

typedef __attribute__((ext_vector_type(16))) _Float16 v16h;
typedef __attribute__((ext_vector_type(8)))  _Float16 v8h;
typedef __attribute__((ext_vector_type(8)))  float    v8f;

#define TPB 256
#define WPB 8   // waves per block (wave32)

// ---- WMMA helpers -------------------------------------------------------
// A-fragment slot order (16-bit A, 16x32): lane<16: slots0-15=k0-15
//                                          lane>=16: slots0-15=k16-31
// LDS tile layout: 16 rows (m) x 32 halves (k), row-major, 512 halves per tile.
__device__ __forceinline__ v16h load_A(const _Float16* tile, int lane) {
  int m    = lane & 15;
  int koff = (lane & 16) ? 16 : 0;
  const _Float16* row = tile + m * 32 + koff;
  v8h lo = *(const v8h*)(row);        // k = koff .. koff+7
  v8h hi = *(const v8h*)(row + 8);    // k = koff+8 .. koff+15
  v16h a;
#pragma unroll
  for (int i = 0; i < 8; ++i) { a[i] = lo[i]; a[i + 8] = hi[i]; }
  return a;
}

__device__ __forceinline__ v8f wmma16(v16h a, v16h b, v8f c) {
  return __builtin_amdgcn_wmma_f32_16x16x32_f16(false, a, false, b, (short)0, c, false, false);
}

// ---- pack x (f32 NCHW, 5ch 9x9) -> a0 f16 [g][11*11 halo][16][8ch pad] ---
__global__ void k_pack_x(const float* __restrict__ x, _Float16* __restrict__ a0, int NG) {
  int total = NG * 121 * 16 * 8;
  for (int i = blockIdx.x * blockDim.x + threadIdx.x; i < total; i += gridDim.x * blockDim.x) {
    int c  = i & 7;
    int s  = (i >> 3) & 15;
    int ph = (i >> 7) % 121;
    int g  = (i >> 7) / 121;
    int y = ph / 11, xx = ph % 11;
    float v = 0.f;
    if (c < 5 && y >= 1 && y <= 9 && xx >= 1 && xx <= 9) {
      int n = g * 16 + s;
      v = x[(((size_t)n * 5 + c) * 9 + (y - 1)) * 9 + (xx - 1)];
    }
    a0[i] = (_Float16)v;
  }
}

// ---- conv1: 3x3, Cin=8(padded from 5), Cout=16, 4 taps per K=32 WMMA ----
// in: a0 haloed 11x11x16x8 ; out: a1 [g][81][16][16], bias+relu
__global__ void k_conv1(const _Float16* __restrict__ a0, const float* __restrict__ Wg,
                        const float* __restrict__ Bg, _Float16* __restrict__ dst, int NG) {
  __shared__ _Float16 sm[3 * 512];
  int tid = threadIdx.x;
  for (int i = tid; i < 3 * 512; i += TPB) {
    int k = i & 31, m = (i >> 5) & 15, q = i >> 9;
    int qi = k >> 3, ci = k & 7, tap = q * 4 + qi;
    float w = 0.f;
    if (tap < 9 && ci < 5) {
      int ky = tap / 3, kx = tap % 3;
      w = Wg[((m * 5 + ci) * 3 + ky) * 3 + kx];
    }
    sm[i] = (_Float16)w;
  }
  __syncthreads();
  int wave = blockIdx.x * WPB + (tid >> 5);
  int lane = tid & 31;
  if (wave >= NG * 81) return;
  int g = wave / 81, p = wave % 81;
  int y = p / 9, x = p % 9, s = lane & 15;
  v8f acc; 
#pragma unroll
  for (int r = 0; r < 8; ++r) acc[r] = 0.f;
#pragma unroll
  for (int q = 0; q < 3; ++q) {
    int t0 = q * 4 + ((lane & 16) ? 2 : 0);
    int t1 = t0 + 1;
    int ta = t0 > 8 ? 8 : t0, tb = t1 > 8 ? 8 : t1;       // clamp addr; zero-A kills garbage
    int pa = (y + ta / 3) * 11 + (x + ta % 3);            // haloed coords: y+1+(dy=-1..1)
    int pb = (y + tb / 3) * 11 + (x + tb % 3);
    v8h b0 = *(const v8h*)(a0 + (((size_t)g * 121 + pa) * 16 + s) * 8);
    v8h b1 = *(const v8h*)(a0 + (((size_t)g * 121 + pb) * 16 + s) * 8);
    v16h b;
#pragma unroll
    for (int i = 0; i < 8; ++i) { b[i] = b0[i]; b[i + 8] = b1[i]; }
    acc = wmma16(load_A(sm + q * 512, lane), b, acc);
  }
  int cobase = (lane & 16) ? 8 : 0;
  v8h h;
#pragma unroll
  for (int r = 0; r < 8; ++r) {
    float v = acc[r] + Bg[cobase + r];
    v = v > 0.f ? v : 0.f;
    h[r] = (_Float16)v;
  }
  *(v8h*)(dst + (((size_t)g * 81 + p) * 16 + s) * 16 + cobase) = h;
}

// ---- generic 1x1 conv (+optional concat src, +nearest upsample, +relu) ---
// src layout [g][Hin*Hin][16][C*]; out [g][PoutT][16][Cout] (halo border = 0)
__global__ void k_conv1x1(const _Float16* __restrict__ src0, const _Float16* __restrict__ src1,
                          int Cs0, int Cs1, int Cin,
                          const float* __restrict__ Wg, const float* __restrict__ Bg,
                          int Cout, _Float16* __restrict__ dst,
                          int Hin, int Hout, int haloOut, int NG) {
  int KC  = (Cin + 31) / 32;
  int CoT = (Cout + 15) / 16;
  __shared__ _Float16 sm[8 * 512];
  int tid = threadIdx.x;
  int nt = CoT * KC;
  for (int i = tid; i < nt * 512; i += TPB) {
    int k = i & 31, m = (i >> 5) & 15, t = i >> 9;
    int kc = t % KC, cot = t / KC;
    int ci = kc * 32 + k, co = cot * 16 + m;
    float w = (co < Cout && ci < Cin) ? Wg[co * Cin + ci] : 0.f;
    sm[i] = (_Float16)w;
  }
  __syncthreads();
  int PoutT = haloOut ? (Hout + 2) * (Hout + 2) : Hout * Hout;
  int wave = blockIdx.x * WPB + (tid >> 5);
  int lane = tid & 31;
  if (wave >= NG * PoutT) return;
  int g = wave / PoutT, pl = wave % PoutT;
  int s = lane & 15;
  int chalf = (lane & 16) ? 8 : 0;
  _Float16* dbase = dst + (((size_t)g * PoutT + pl) * 16 + s) * Cout + chalf;
  int y, x; bool border = false;
  if (haloOut) {
    int hy = pl / (Hout + 2), hx = pl % (Hout + 2);
    border = (hy == 0) || (hy == Hout + 1) || (hx == 0) || (hx == Hout + 1);
    y = hy - 1; x = hx - 1;
  } else { y = pl / Hout; x = pl % Hout; }
  if (border) {
    v8h z;
#pragma unroll
    for (int r = 0; r < 8; ++r) z[r] = (_Float16)0.f;
#pragma unroll
    for (int cot = 0; cot < 4; ++cot) if (cot < CoT) *(v8h*)(dbase + cot * 16) = z;
    return;
  }
  int my = (y * Hin) / Hout, mx = (x * Hin) / Hout;   // nearest upsample (id when Hin==Hout)
  int pin = my * Hin + mx;
  v16h bf[2];
#pragma unroll
  for (int kc = 0; kc < 2; ++kc) if (kc < KC) {
    int c0 = kc * 32 + ((lane & 16) ? 16 : 0);
    bool ovf = c0 >= Cin;            // Cin==16: lanes>=16 read garbage, A is zero there
    bool in1 = (!ovf) && (c0 >= Cs0);
    const _Float16* sp = in1 ? src1 : src0;
    int C_ = in1 ? Cs1 : Cs0;
    int cl = ovf ? 0 : (in1 ? (c0 - Cs0) : c0);
    bf[kc] = *(const v16h*)(sp + (((size_t)g * Hin * Hin + pin) * 16 + s) * C_ + cl);
  }
#pragma unroll
  for (int cot = 0; cot < 4; ++cot) if (cot < CoT) {
    v8f acc;
#pragma unroll
    for (int r = 0; r < 8; ++r) acc[r] = 0.f;
#pragma unroll
    for (int kc = 0; kc < 2; ++kc) if (kc < KC)
      acc = wmma16(load_A(sm + (cot * KC + kc) * 512, lane), bf[kc], acc);
    int cobase = cot * 16 + chalf;
    v8h h;
#pragma unroll
    for (int r = 0; r < 8; ++r) {
      float v = acc[r] + Bg[cobase + r];
      v = v > 0.f ? v : 0.f;
      h[r] = (_Float16)v;
    }
    *(v8h*)(dbase + cot * 16) = h;
  }
}

// ---- 3x3 conv, Cin=16 (two taps per WMMA), Cout=32, raw(bias, no relu) ---
// src: t1 haloed [g][121][16][16]; out t2 [g][81][16][32]
__global__ void k_conv3x3_c16(const _Float16* __restrict__ src, const float* __restrict__ Wg,
                              const float* __restrict__ Bg, _Float16* __restrict__ dst, int NG) {
  __shared__ _Float16 sm[10 * 512];   // [cot(2)][pair(5)] tiles
  int tid = threadIdx.x;
  for (int i = tid; i < 10 * 512; i += TPB) {
    int k = i & 31, m = (i >> 5) & 15, t = i >> 9;
    int pr = t % 5, cot = t / 5;
    int tap = 2 * pr + (k >> 4), ci = k & 15;
    int co = cot * 16 + m;
    float w = 0.f;
    if (tap < 9) {
      int ky = tap / 3, kx = tap % 3;
      w = Wg[((co * 16 + ci) * 3 + ky) * 3 + kx];
    }
    sm[i] = (_Float16)w;
  }
  __syncthreads();
  int wave = blockIdx.x * WPB + (tid >> 5);
  int lane = tid & 31;
  if (wave >= NG * 81) return;
  int g = wave / 81, p = wave % 81;
  int y = p / 9, x = p % 9, s = lane & 15;
  v8f acc[2];
#pragma unroll
  for (int c = 0; c < 2; ++c)
#pragma unroll
    for (int r = 0; r < 8; ++r) acc[c][r] = 0.f;
#pragma unroll
  for (int pr = 0; pr < 5; ++pr) {
    int tap = 2 * pr + ((lane & 16) ? 1 : 0);
    int tc = tap > 8 ? 8 : tap;
    int pin = (y + tc / 3) * 11 + (x + tc % 3);
    v16h b = *(const v16h*)(src + (((size_t)g * 121 + pin) * 16 + s) * 16);
#pragma unroll
    for (int cot = 0; cot < 2; ++cot)
      acc[cot] = wmma16(load_A(sm + (cot * 5 + pr) * 512, lane), b, acc[cot]);
  }
  int chalf = (lane & 16) ? 8 : 0;
#pragma unroll
  for (int cot = 0; cot < 2; ++cot) {
    int cobase = cot * 16 + chalf;
    v8h h;
#pragma unroll
    for (int r = 0; r < 8; ++r) h[r] = (_Float16)(acc[cot][r] + Bg[cobase + r]);
    *(v8h*)(dst + (((size_t)g * 81 + p) * 16 + s) * 32 + cobase) = h;
  }
}

// ---- generic 3x3 conv, Cin in {32,64}, haloed or tap-skip input ----------
__global__ void k_conv3x3(const _Float16* __restrict__ src, int Cin,
                          const float* __restrict__ Wg, const float* __restrict__ Bg,
                          int Cout, _Float16* __restrict__ dst,
                          int H, int haloIn, int doRelu, int NG) {
  int KC = Cin / 32, CoT = Cout / 16;
  __shared__ _Float16 sm[36 * 512];
  int tid = threadIdx.x;
  int nt = CoT * 9 * KC;
  for (int i = tid; i < nt * 512; i += TPB) {
    int k = i & 31, m = (i >> 5) & 15, t = i >> 9;
    int kc = t % KC, tap = (t / KC) % 9, cot = t / (KC * 9);
    int ci = kc * 32 + k, co = cot * 16 + m;
    int ky = tap / 3, kx = tap % 3;
    sm[i] = (_Float16)Wg[((co * Cin + ci) * 3 + ky) * 3 + kx];
  }
  __syncthreads();
  int wave = blockIdx.x * WPB + (tid >> 5);
  int lane = tid & 31;
  if (wave >= NG * H * H) return;
  int g = wave / (H * H), p = wave % (H * H);
  int y = p / H, x = p % H, s = lane & 15;
  int Win = haloIn ? H + 2 : H;
  int Pin = Win * Win;
  v8f acc[4];
#pragma unroll
  for (int c = 0; c < 4; ++c)
#pragma unroll
    for (int r = 0; r < 8; ++r) acc[c][r] = 0.f;
  for (int tap = 0; tap < 9; ++tap) {
    int dy = tap / 3 - 1, dx = tap % 3 - 1;
    int py, px;
    if (haloIn) { py = y + 1 + dy; px = x + 1 + dx; }
    else { py = y + dy; px = x + dx; if (py < 0 || py >= H || px < 0 || px >= H) continue; }
    int pin = py * Win + px;
    const _Float16* pb = src + (((size_t)g * Pin + pin) * 16 + s) * Cin + ((lane & 16) ? 16 : 0);
#pragma unroll
    for (int kc = 0; kc < 2; ++kc) if (kc < KC) {
      v16h b = *(const v16h*)(pb + kc * 32);
#pragma unroll
      for (int cot = 0; cot < 4; ++cot) if (cot < CoT)
        acc[cot] = wmma16(load_A(sm + ((cot * 9 + tap) * KC + kc) * 512, lane), b, acc[cot]);
    }
  }
  int chalf = (lane & 16) ? 8 : 0;
#pragma unroll
  for (int cot = 0; cot < 4; ++cot) if (cot < CoT) {
    int cobase = cot * 16 + chalf;
    v8h h;
#pragma unroll
    for (int r = 0; r < 8; ++r) {
      float v = acc[cot][r] + Bg[cobase + r];
      if (doRelu) v = v > 0.f ? v : 0.f;
      h[r] = (_Float16)v;
    }
    *(v8h*)(dst + (((size_t)g * H * H + p) * 16 + s) * Cout + cobase) = h;
  }
}

// ---- maxpool kxk stride1 VALID + relu ------------------------------------
__global__ void k_pool(const _Float16* __restrict__ src, _Float16* __restrict__ dst,
                       int C, int Hin, int K, int NG) {
  int Hout = Hin - K + 1;
  int CQ = C / 8;
  int total = NG * Hout * Hout * 16 * CQ;
  for (int i = blockIdx.x * blockDim.x + threadIdx.x; i < total; i += gridDim.x * blockDim.x) {
    int cq = i % CQ;
    int s = (i / CQ) & 15;
    int p = (i / (CQ * 16)) % (Hout * Hout);
    int g = i / (CQ * 16 * Hout * Hout);
    int y = p / Hout, x = p % Hout;
    v8h m;
#pragma unroll
    for (int r = 0; r < 8; ++r) m[r] = (_Float16)(-65504.f);
    for (int dy = 0; dy < K; ++dy)
      for (int dx = 0; dx < K; ++dx) {
        int pin = (y + dy) * Hin + (x + dx);
        v8h v = *(const v8h*)(src + (((size_t)g * Hin * Hin + pin) * 16 + s) * C + cq * 8);
#pragma unroll
        for (int r = 0; r < 8; ++r) m[r] = v[r] > m[r] ? v[r] : m[r];
      }
#pragma unroll
    for (int r = 0; r < 8; ++r) { _Float16 z = (_Float16)0.f; m[r] = m[r] > z ? m[r] : z; }
    *(v8h*)(dst + (((size_t)g * Hout * Hout + p) * 16 + s) * C + cq * 8) = m;
  }
}

// ---- final 1x1: concat(out0, u2b) 32ch -> 2ch, f32 NCHW output -----------
__global__ void k_final(const _Float16* __restrict__ src0, const _Float16* __restrict__ src1,
                        const float* __restrict__ Wg, const float* __restrict__ Bg,
                        float* __restrict__ out, int NG) {
  __shared__ _Float16 sm[512];
  int tid = threadIdx.x;
  for (int i = tid; i < 512; i += TPB) {
    int k = i & 31, m = i >> 5;
    sm[i] = (_Float16)((m < 2) ? Wg[m * 32 + k] : 0.f);
  }
  __syncthreads();
  int wave = blockIdx.x * WPB + (tid >> 5);
  int lane = tid & 31;
  if (wave >= NG * 81) return;
  int g = wave / 81, p = wave % 81, s = lane & 15;
  const _Float16* sp = (lane & 16) ? src1 : src0;   // k0-15 = out0, k16-31 = u2b
  v16h b = *(const v16h*)(sp + (((size_t)g * 81 + p) * 16 + s) * 16);
  v8f acc;
#pragma unroll
  for (int r = 0; r < 8; ++r) acc[r] = 0.f;
  acc = wmma16(load_A(sm, lane), b, acc);
  if (lane < 16) {
    size_t n = (size_t)g * 16 + lane;
    out[(n * 2 + 0) * 81 + p] = acc[0] + Bg[0];
    out[(n * 2 + 1) * 81 + p] = acc[1] + Bg[1];
  }
}

extern "C" void kernel_launch(void* const* d_in, const int* in_sizes, int n_in,
                              void* d_out, int out_size, void* d_ws, size_t ws_size,
                              hipStream_t stream) {
  const float* x    = (const float*)d_in[0];
  const float* iw1  = (const float*)d_in[1];  const float* ib1  = (const float*)d_in[2];
  const float* iw2  = (const float*)d_in[3];  const float* ib2  = (const float*)d_in[4];
  const float* d1w1 = (const float*)d_in[5];  const float* d1b1 = (const float*)d_in[6];
  const float* d1w2 = (const float*)d_in[7];  const float* d1b2 = (const float*)d_in[8];
  const float* d2w1 = (const float*)d_in[9];  const float* d2b1 = (const float*)d_in[10];
  const float* d2w2 = (const float*)d_in[11]; const float* d2b2 = (const float*)d_in[12];
  const float* u1w1 = (const float*)d_in[13]; const float* u1b1 = (const float*)d_in[14];
  const float* u1w2 = (const float*)d_in[15]; const float* u1b2 = (const float*)d_in[16];
  const float* u2w1 = (const float*)d_in[17]; const float* u2b1 = (const float*)d_in[18];
  const float* u2w2 = (const float*)d_in[19]; const float* u2b2 = (const float*)d_in[20];
  const float* fw   = (const float*)d_in[21]; const float* fb   = (const float*)d_in[22];

  int N  = in_sizes[0] / 405;   // B = 16384
  int NG = N / 16;              // 16-sample groups

  auto al = [](size_t v) { return (v + 255) & ~(size_t)255; };
  char* ws = (char*)d_ws;
  size_t szOut0 = al((size_t)NG * 81  * 16 * 16 * 2);
  size_t szD1   = al((size_t)NG * 25  * 16 * 32 * 2);
  size_t szU1B  = al((size_t)NG * 25  * 16 * 32 * 2);
  size_t szA0   = al((size_t)NG * 121 * 16 * 8  * 2);
  size_t szT1   = al((size_t)NG * 121 * 16 * 16 * 2);
  size_t szT3   = al((size_t)NG * 49  * 16 * 32 * 2);
  size_t szD2   = al((size_t)NG * 9   * 16 * 64 * 2);
  size_t szT6   = al((size_t)NG * 81  * 16 * 64 * 2);

  _Float16* OUT0 = (_Float16*)(ws);
  _Float16* D1   = (_Float16*)(ws + szOut0);
  _Float16* U1B  = (_Float16*)(ws + szOut0 + szD1);
  char* S = ws + szOut0 + szD1 + szU1B;          // reused scratch slab (lifetimes disjoint)
  _Float16* A0  = (_Float16*)S;  _Float16* A1  = (_Float16*)(S + szA0);
  _Float16* T1  = (_Float16*)S;  _Float16* T2  = (_Float16*)(S + szT1);
  _Float16* T3  = (_Float16*)S;  _Float16* T4  = (_Float16*)(S + szT3);
  _Float16* D2  = (_Float16*)S;  _Float16* T5  = (_Float16*)(S + szD2);
  _Float16* T6  = (_Float16*)S;  _Float16* U2B = (_Float16*)(S + szT6);

  dim3 blk(TPB);
  auto nb = [](int waves) { return dim3((unsigned)((waves + WPB - 1) / WPB)); };
  dim3 ew(2048);

  k_pack_x<<<ew, blk, 0, stream>>>(x, A0, NG);
  k_conv1<<<nb(NG * 81), blk, 0, stream>>>(A0, iw1, ib1, A1, NG);
  // iw2: 1x1 16->16, relu -> out0
  k_conv1x1<<<nb(NG * 81),  blk, 0, stream>>>(A1, A1, 16, 16, 16, iw2,  ib2,  16, OUT0, 9, 9, 0, NG);
  // d1w1: 1x1 16->16, relu -> t1 (haloed 11x11)
  k_conv1x1<<<nb(NG * 121), blk, 0, stream>>>(OUT0, OUT0, 16, 16, 16, d1w1, d1b1, 16, T1, 9, 9, 1, NG);
  // d1w2: 3x3 16->32 (raw) -> t2
  k_conv3x3_c16<<<nb(NG * 81), blk, 0, stream>>>(T1, d1w2, d1b2, T2, NG);
  k_pool<<<ew, blk, 0, stream>>>(T2, D1, 32, 9, 5, NG);
  // d2w1: 1x1 32->32, relu -> t3 (haloed 7x7)
  k_conv1x1<<<nb(NG * 49), blk, 0, stream>>>(D1, D1, 32, 32, 32, d2w1, d2b1, 32, T3, 5, 5, 1, NG);
  // d2w2: 3x3 32->64 (raw) -> t4
  k_conv3x3<<<nb(NG * 25), blk, 0, stream>>>(T3, 32, d2w2, d2b2, 64, T4, 5, 1, 0, NG);
  k_pool<<<ew, blk, 0, stream>>>(T4, D2, 64, 5, 3, NG);
  // u1w1: upsample 3->5 + 1x1 64->64, relu -> t5 (haloed 7x7)
  k_conv1x1<<<nb(NG * 49), blk, 0, stream>>>(D2, D2, 64, 64, 64, u1w1, u1b1, 64, T5, 3, 5, 1, NG);
  // u1w2: 3x3 64->32, relu -> u1b
  k_conv3x3<<<nb(NG * 25), blk, 0, stream>>>(T5, 64, u1w2, u1b2, 32, U1B, 5, 1, 1, NG);
  // u2w1: upsample 5->9 + concat(d1,u1b) + 1x1 64->64, relu -> t6 (no halo)
  k_conv1x1<<<nb(NG * 81), blk, 0, stream>>>(D1, U1B, 32, 32, 64, u2w1, u2b1, 64, T6, 5, 9, 0, NG);
  // u2w2: 3x3 64->16 (tap-skip zero pad), relu -> u2b
  k_conv3x3<<<nb(NG * 81), blk, 0, stream>>>(T6, 64, u2w2, u2b2, 16, U2B, 9, 0, 1, NG);
  // final: 1x1 concat(out0,u2b) 32->2, f32 NCHW
  k_final<<<nb(NG * 81), blk, 0, stream>>>(OUT0, U2B, fw, fb, (float*)d_out, NG);
}
